// HierarchicalQuantizer_89781996355991
// MI455X (gfx1250) — matrix operand order repaired and checked by reading, not compiled, not run on hardware
//
#include <hip/hip_runtime.h>
#include <hip/hip_bf16.h>
#include <math.h>

// Problem constants (fixed by the reference): x[64,1024,256] f32, codebook[1024,256] f32
#define NROW 65536      // 64*1024 flattened rows
#define DDIM 256
#define KCODE 1024
#define QELEMS (64ull * 1024ull * 256ull)   // 16,777,216 quantized_st elements

#define CHUNK_CODES 64                        // codes staged per LDS chunk
#define CHUNK_BYTES (CHUNK_CODES * DDIM * 2)  // 32 KB (f16)
#define NCHUNK (KCODE / CHUNK_CODES)          // 16

typedef __attribute__((ext_vector_type(16))) _Float16 v16h;
typedef __attribute__((ext_vector_type(8)))  float    v8f;

// Low 32 bits of a generic pointer to LDS == wave-relative LDS byte offset
__device__ __forceinline__ unsigned lds_off_of(const void* p) {
    return (unsigned)(unsigned long long)p;
}

// ---------------------------------------------------------------------------
// Kernel A: codebook f32 -> f16 copy + per-code squared norms (one block/code)
// ---------------------------------------------------------------------------
__global__ void cb_convert_kernel(const float* __restrict__ cb,
                                  _Float16* __restrict__ cbh,
                                  float* __restrict__ cnorm) {
    int k = blockIdx.x;            // 0..1023
    int t = threadIdx.x;           // 0..255
    float v = cb[k * DDIM + t];
    cbh[k * DDIM + t] = (_Float16)v;
    float s = v * v;
    #pragma unroll
    for (int m = 16; m >= 1; m >>= 1) s += __shfl_xor(s, m, 32);
    __shared__ float red[8];
    int wave = t >> 5;
    if ((t & 31) == 0) red[wave] = s;
    __syncthreads();
    if (t == 0) {
        float tot = 0.f;
        #pragma unroll
        for (int w = 0; w < 8; ++w) tot += red[w];
        cnorm[k] = tot;
    }
}

// ---------------------------------------------------------------------------
// Kernel B (fused): WMMA distance GEMM + argmin + gather + straight-through
// output + MSE partials. Codebook chunks double-buffered in LDS via
// global_load_async_to_lds_b128 (ASYNCcnt), shared by all 8 waves.
// ---------------------------------------------------------------------------
__global__ __launch_bounds__(256)
void vq_fused_kernel(const float* __restrict__ x,
                     const _Float16* __restrict__ cbh,
                     const float* __restrict__ cnorm,
                     const float* __restrict__ cb32,
                     int* __restrict__ idx_out,
                     float* __restrict__ idx_f_out,
                     float* __restrict__ out_q,
                     float* __restrict__ mse_accum) {
    __shared__ _Float16 smem[2 * CHUNK_CODES * DDIM];   // 64 KB double buffer

    const int lane = threadIdx.x & 31;
    const int wave = threadIdx.x >> 5;
    const int hi   = lane >> 4;      // half-wave select (0/1)
    const int sub  = lane & 15;      // lane within half
    const int row_base = blockIdx.x * 128 + wave * 16;
    const int arow = row_base + sub; // A-matrix row this lane feeds

    // --- async stage of one 32KB codebook chunk: 256 thr x 8 x 16B ---------
    const char* cbh_bytes = (const char*)cbh;
    const unsigned smem_base = lds_off_of(smem);
    auto issue_chunk = [&](int chunk, int bufsel) {
        const char* g = cbh_bytes + (size_t)chunk * CHUNK_BYTES + threadIdx.x * 16;
        unsigned l = smem_base + (unsigned)bufsel * CHUNK_BYTES + threadIdx.x * 16;
        #pragma unroll
        for (int j = 0; j < 8; ++j) {
            asm volatile("global_load_async_to_lds_b128 %0, %1, off"
                         :: "v"(l + j * 4096u), "v"(g + (size_t)j * 4096)
                         : "memory");
        }
    };

    // --- Load + convert A fragments (f32 -> f16), contiguous k order (not the ISA table order):
// lane<16 holds K = 0..15 (+32*kb); lane>=16 holds K = 16..31 (matches B).
    issue_chunk(0, 0);   // overlap chunk-0 fill with A-fragment conversion

    v16h a[8];
    const float4* xr4 = (const float4*)(x + (size_t)arow * DDIM);
    #pragma unroll
    for (int kb = 0; kb < 8; ++kb) {
        int q0 = (kb * 32 + hi * 16) >> 2;     // float4 index
        float4 f0 = xr4[q0],     f1 = xr4[q0 + 1];
        float4 f2 = xr4[q0 + 2], f3 = xr4[q0 + 3];   // +8 dims
        union { v16h v; _Float16 h[16]; } u;
        u.h[0]  = (_Float16)f0.x; u.h[1]  = (_Float16)f0.y;
        u.h[2]  = (_Float16)f0.z; u.h[3]  = (_Float16)f0.w;
        u.h[4]  = (_Float16)f1.x; u.h[5]  = (_Float16)f1.y;
        u.h[6]  = (_Float16)f1.z; u.h[7]  = (_Float16)f1.w;
        u.h[8]  = (_Float16)f2.x; u.h[9]  = (_Float16)f2.y;
        u.h[10] = (_Float16)f2.z; u.h[11] = (_Float16)f2.w;
        u.h[12] = (_Float16)f3.x; u.h[13] = (_Float16)f3.y;
        u.h[14] = (_Float16)f3.z; u.h[15] = (_Float16)f3.w;
        a[kb] = u.v;
    }

    float best_v[8];
    int   best_i[8];
    #pragma unroll
    for (int r = 0; r < 8; ++r) { best_v[r] = 3.4e38f; best_i[r] = 0; }

    asm volatile("s_wait_asynccnt 0x0" ::: "memory");
    __syncthreads();                                   // chunk 0 ready

    #pragma unroll 1
    for (int c = 0; c < NCHUNK; ++c) {
        if (c + 1 < NCHUNK) issue_chunk(c + 1, (c + 1) & 1);
        const _Float16* cur = smem + (c & 1) * (CHUNK_CODES * DDIM);

        #pragma unroll 1
        for (int tt = 0; tt < CHUNK_CODES / 16; ++tt) {
            const int code_local = tt * 16 + sub;          // B column in chunk
            const int code = c * CHUNK_CODES + code_local; // global code id
            float cn = cnorm[code];   // issue early; covered by ds+wmma below

            // B 32x16 in contiguous k order (same as A): lane<16 holds K=0..15, lane>=16 K=16..31
            // -> 16 contiguous f16 of the codebook row (from LDS).
            // Preload ALL tile fragments so the 16 ds_load_b128 issue
            // back-to-back and the 8 WMMAs then chain without per-frag waits.
            const _Float16* wr = cur + (size_t)code_local * DDIM + hi * 16;
            v16h bf[8];
            #pragma unroll
            for (int kb = 0; kb < 8; ++kb)
                bf[kb] = *(const v16h*)(wr + kb * 32);

            v8f acc = {};
            #pragma unroll
            for (int kb = 0; kb < 8; ++kb)
                acc = __builtin_amdgcn_wmma_f32_16x16x32_f16(
                          false, a[kb], false, bf[kb], (short)0, acc,
                          false, false);

            #pragma unroll
            for (int r = 0; r < 8; ++r) {
                float s = __builtin_fmaf(-2.0f, acc[r], cn);
                if (s < best_v[r]) { best_v[r] = s; best_i[r] = code; }
            }
        }
        if (c + 1 < NCHUNK) {
            asm volatile("s_wait_asynccnt 0x0" ::: "memory");
            __syncthreads();                           // publish chunk c+1
        }
    }

    // --- argmin across the 16 columns in each half-wave (ties -> lower index,
    // matching jnp.argmin). Masks 1..8 keep lanes within their 16-lane half.
    #pragma unroll
    for (int m = 1; m <= 8; m <<= 1) {
        #pragma unroll
        for (int r = 0; r < 8; ++r) {
            float ov = __shfl_xor(best_v[r], m, 32);
            int   oi = __shfl_xor(best_i[r], m, 32);
            if (ov < best_v[r] || (ov == best_v[r] && oi < best_i[r])) {
                best_v[r] = ov; best_i[r] = oi;
            }
        }
    }
    // C layout: lanes 0-15 hold rows M=r (VGPR r), lanes 16-31 hold M=8+r.
    if (sub == 0) {
        #pragma unroll
        for (int r = 0; r < 8; ++r) {
            int row = row_base + hi * 8 + r;
            idx_out[row]   = best_i[r];
            idx_f_out[row] = (float)best_i[r];
        }
    }

    // --- fused epilogue: gather fp32 codebook row, straight-through output,
    // MSE partial. x rows were just read -> hot in WGP$/L2.
    float msum = 0.f;
    #pragma unroll
    for (int r = 0; r < 16; ++r) {
        int kidx = __shfl(best_i[r & 7], (r & 8) ? 16 : 0, 32);
        int row  = row_base + r;
        const float4* xr = (const float4*)(x    + (size_t)row  * DDIM) + lane * 2;
        const float4* qr = (const float4*)(cb32 + (size_t)kidx * DDIM) + lane * 2;
        float4*       orp = (float4*)(out_q + (size_t)row * DDIM) + lane * 2;
        float4 xa = xr[0], xb = xr[1];
        float4 qa = qr[0], qb = qr[1];
        float4 oa, ob;
        #define ST_COMP(dst, xv, qv, comp) { float d_ = qv.comp - xv.comp; \
                dst.comp = xv.comp + d_; msum = __builtin_fmaf(d_, d_, msum); }
        ST_COMP(oa, xa, qa, x) ST_COMP(oa, xa, qa, y)
        ST_COMP(oa, xa, qa, z) ST_COMP(oa, xa, qa, w)
        ST_COMP(ob, xb, qb, x) ST_COMP(ob, xb, qb, y)
        ST_COMP(ob, xb, qb, z) ST_COMP(ob, xb, qb, w)
        #undef ST_COMP
        orp[0] = oa; orp[1] = ob;
    }
    #pragma unroll
    for (int m = 16; m >= 1; m >>= 1) msum += __shfl_xor(msum, m, 32);
    if (lane == 0) atomicAdd(mse_accum, msum);
}

// ---------------------------------------------------------------------------
// Kernel D: LDS-privatized histogram of code usage (64 blocks x 256 thr x 4).
// ---------------------------------------------------------------------------
__global__ void vq_hist_kernel(const int* __restrict__ idx,
                               int* __restrict__ counts) {
    __shared__ int h[KCODE];
    for (int i = threadIdx.x; i < KCODE; i += 256) h[i] = 0;
    __syncthreads();
    int base = blockIdx.x * 1024;
    #pragma unroll
    for (int j = 0; j < 4; ++j)
        atomicAdd(&h[idx[base + j * 256 + threadIdx.x]], 1);
    __syncthreads();
    for (int i = threadIdx.x; i < KCODE; i += 256)
        if (h[i]) atomicAdd(&counts[i], h[i]);
}

// ---------------------------------------------------------------------------
// Kernel E: scalars — vq_total_loss = 1.25 * mse, perplexity from histogram.
// ---------------------------------------------------------------------------
__global__ void vq_final_kernel(const int* __restrict__ counts,
                                const float* __restrict__ mse_accum,
                                float* __restrict__ out_scalars) {
    int t = threadIdx.x;               // 1024 threads, one per code
    float p = (float)counts[t] / (float)NROW;
    float term = p * logf(p + 1e-10f);
    #pragma unroll
    for (int m = 16; m >= 1; m >>= 1) term += __shfl_xor(term, m, 32);
    __shared__ float red[32];
    int wave = t >> 5;
    if ((t & 31) == 0) red[wave] = term;
    __syncthreads();
    if (t == 0) {
        float s = 0.f;
        #pragma unroll
        for (int w = 0; w < 32; ++w) s += red[w];
        float mse = (*mse_accum) / (float)QELEMS;
        out_scalars[0] = 1.25f * mse;  // codebook_loss + 0.25*commitment (equal)
        out_scalars[1] = expf(-s);
    }
}

// ---------------------------------------------------------------------------
extern "C" void kernel_launch(void* const* d_in, const int* in_sizes, int n_in,
                              void* d_out, int out_size, void* d_ws, size_t ws_size,
                              hipStream_t stream) {
    (void)in_sizes; (void)n_in; (void)out_size; (void)ws_size;
    const float* x  = (const float*)d_in[0];
    const float* cb = (const float*)d_in[1];

    float* out        = (float*)d_out;
    float* out_q      = out;                   // [QELEMS]
    float* out_scal   = out + QELEMS;          // loss, perplexity
    float* out_idx_f  = out + QELEMS + 2;      // [NROW] indices as float

    // Workspace layout (all offsets 256B-aligned): ~800 KB total
    char* ws = (char*)d_ws;
    _Float16* cbh   = (_Float16*)ws;                       // 512 KB
    float*    cnorm = (float*)(ws + 524288);               // 4 KB
    int*      idxb  = (int*)  (ws + 524288 + 4096);        // 256 KB
    int*      cnt   = (int*)  (ws + 524288 + 4096 + 262144); // 4 KB
    float*    msea  = (float*)(ws + 524288 + 4096 + 262144 + 4096);

    // zero histogram + mse accumulator (contiguous region)
    hipMemsetAsync(cnt, 0, 4096 + 256, stream);

    cb_convert_kernel<<<KCODE, 256, 0, stream>>>(cb, cbh, cnorm);
    vq_fused_kernel<<<NROW / 128, 256, 0, stream>>>(x, cbh, cnorm, cb,
                                                    idxb, out_idx_f, out_q, msea);
    vq_hist_kernel<<<NROW / 1024, 256, 0, stream>>>(idxb, cnt);
    vq_final_kernel<<<1, KCODE, 0, stream>>>(cnt, msea, out_scal);
}
